// DeepAR_25417616458035
// MI455X (gfx1250) — matrix operand order repaired and checked by reading, not compiled, not run on hardware
//
#include <hip/hip_runtime.h>
#include <math.h>

// ---------------------------------------------------------------------------
// DeepAR 2-layer LSTM (B=512, H=512, F=64, TO=128, TP=192) for gfx1250.
// Core GEMMs run on v_wmma_f32_16x16x32_f16 with fp32 accumulation and a
// fully fused LSTM-cell epilogue (sigmoid/tanh + c/h update in registers).
// Waves within a block share identical weight-tile addresses so the WGP$/L0
// dedupes the L2 weight stream (weights are L2-resident: 6.5 MB f16), and
// WGP-scope global_prefetch_b8 warms L0 two K-iterations ahead.
// ---------------------------------------------------------------------------

typedef __attribute__((ext_vector_type(16))) _Float16 v16h;
typedef __attribute__((ext_vector_type(8)))  _Float16 v8h;
typedef __attribute__((ext_vector_type(8)))  float    v8f;

#define B_    512
#define TP_   192
#define TO_   128
#define F_    64
#define H_    512
#define K1PAD 608            // F+1+H = 577 padded to 19*32
#define KT1   19
#define K2_   1024
#define KT2   32
#define NT_   128            // 2048 / 16 N-tiles

__device__ __forceinline__ float sigmoidf_(float x) {
    return 1.0f / (1.0f + __expf(-x));
}

// ---------------------------------------------------------------------------
// Weight conversion: fp32 row-major [Ksrc, 2048] -> f16 swizzled tiles.
// Tile (kt, nt) = 32(K) x 16(N); element order is contiguous k, same as the A
// fragment: lane l (<16) holds K = kt*32 + 0..15, lanes 16..31 hold
// K = kt*32 + 16..31, N = nt*16 + (l & 15); each lane's 16 f16 contiguous.
// Grid: KT*128 blocks of 32 threads.
// ---------------------------------------------------------------------------
__global__ void convert_weights_f16(const float* __restrict__ W, int Ksrc,
                                    _Float16* __restrict__ dst)
{
    const int tile = blockIdx.x;          // kt*128 + nt
    const int kt   = tile >> 7;
    const int nt   = tile & 127;
    const int lane = threadIdx.x;
    const int n    = nt * 16 + (lane & 15);
    const int kb   = kt * 32 + ((lane < 16) ? 0 : 16);
    _Float16* d = dst + (size_t)tile * 512 + (size_t)lane * 16;
#pragma unroll
    for (int e = 0; e < 16; ++e) {
        const int k = kb + e;
        const float v = (k < Ksrc) ? W[(size_t)k * 2048 + n] : 0.0f;
        d[e] = (_Float16)v;
    }
}

// ---------------------------------------------------------------------------
// scale[b] = 1 + mean_t y[b,t];  inv_scale = 1/scale.  One wave per row.
// ---------------------------------------------------------------------------
__global__ __launch_bounds__(256) void scale_kernel(const float* __restrict__ y,
                                                    float* __restrict__ scale,
                                                    float* __restrict__ inv_scale)
{
    const int b    = blockIdx.x * (blockDim.x >> 5) + (threadIdx.x >> 5);
    const int lane = threadIdx.x & 31;
    float s = 0.0f;
    for (int t = lane; t < TO_; t += 32) s += y[(size_t)b * TO_ + t];
#pragma unroll
    for (int off = 16; off; off >>= 1) s += __shfl_xor(s, off, 32);
    if (lane == 0) {
        const float sc = 1.0f + s * (1.0f / (float)TO_);
        scale[b]     = sc;
        inv_scale[b] = 1.0f / sc;
    }
}

// ---------------------------------------------------------------------------
// Per-step feature prep: feat1[b, 0..63] = x[b, t, :] (f16).
// Conditioning steps also write feat1[b, 64] = y_sc[t-1, b] (0 at t==0).
// AR steps leave col 64 to the head kernel (predicted mean feedback).
// Grid: 512 blocks x 64 threads.
// ---------------------------------------------------------------------------
__global__ void prep_kernel(const float* __restrict__ x,
                            const float* __restrict__ y,
                            const float* __restrict__ inv_scale,
                            _Float16* __restrict__ feat1, int t, int cond)
{
    const int b = blockIdx.x;
    const int f = threadIdx.x;
    feat1[(size_t)b * K1PAD + f] = (_Float16)x[((size_t)b * TP_ + t) * F_ + f];
    if (cond && f == 0) {
        const float yv = (t == 0) ? 0.0f
                                  : y[(size_t)b * TO_ + (t - 1)] * inv_scale[b];
        feat1[(size_t)b * K1PAD + 64] = (_Float16)yv;
    }
}

// ---------------------------------------------------------------------------
// Fused LSTM cell: gates = feat @ Wsw + bias, then
//   c' = sig(f+1)*c + sig(i)*tanh(g);  h' = sig(o)*tanh(c')
// Each wave: 16 rows x 16 cols across all 4 gates (4 fp32 accumulators).
// Grid: 128 blocks x 256 threads = 1024 waves = 32 M-tiles x 32 col-groups.
// Mapping: mt = wid&31 (varies within block), ng = wid>>5 (uniform within
// block) -> all 8 waves of a block load identical weight tiles (L0 dedup).
// h' written as f16 into next-step feature buffers (and fp32 z for cell 2).
// ---------------------------------------------------------------------------
__global__ __launch_bounds__(256) void lstm_cell_wmma(
    const _Float16* __restrict__ feat, int kpad, int ktiles,
    const _Float16* __restrict__ Wsw, const float* __restrict__ bias,
    float* __restrict__ c,
    _Float16* __restrict__ dA, int dA_stride, int dA_off,
    _Float16* __restrict__ dB, int dB_stride, int dB_off,
    float* __restrict__ zdst)
{
    const int lane = threadIdx.x & 31;
    const int wid  = blockIdx.x * (blockDim.x >> 5) + (threadIdx.x >> 5);
    const int mt   = wid & 31;          // varies across waves of a block
    const int ng   = wid >> 5;          // uniform within a block (8 | 32)
    const int m0   = mt << 4;
    const int half = lane >> 4;         // A/C layout half-select
    const int l16  = lane & 15;

    // A fragment (16x32 f16): lane<16 -> K {0..15}, lane>=16 -> {16..31}
    const _Float16* aptr = feat + (size_t)(m0 + l16) * kpad + (half ? 16 : 0);
    // B fragment: swizzled tiles, gate gi at tile (kt*128 + gi*32 + ng)
    const _Float16* bptr = Wsw + ((size_t)ng * 512) + ((size_t)lane * 16);

    v8f accI = {}; v8f accG = {}; v8f accF = {}; v8f accO = {};

    for (int kt = 0; kt < ktiles; ++kt) {
        // WGP-scope (default scope) prefetch of the 4 gate weight tiles two
        // K-iterations ahead: pulls into all cache levels incl. WGP$/L0,
        // which the 8 B-sharing waves of this block then hit. Unconditional
        // straight-line code keeps the loop software-pipelined; prefetches
        // are speculative and update no counters, and the over-reach past
        // the weight array stays inside the workspace.
        {
            const _Float16* pf = bptr + 2u * (size_t)NT_ * 512u;
            asm volatile("global_prefetch_b8 %0, off\n\t"
                         "global_prefetch_b8 %0, off offset:32768\n\t"
                         "global_prefetch_b8 %0, off offset:65536\n\t"
                         "global_prefetch_b8 %0, off offset:98304"
                         :: "v"(pf));
        }

        union { v16h v; v8h h[2]; } a;
        a.h[0] = *reinterpret_cast<const v8h*>(aptr);
        a.h[1] = *reinterpret_cast<const v8h*>(aptr + 8);
        aptr += 32;

        const v16h b0 = *reinterpret_cast<const v16h*>(bptr);                 // i
        const v16h b1 = *reinterpret_cast<const v16h*>(bptr + 32u * 512u);    // g
        const v16h b2 = *reinterpret_cast<const v16h*>(bptr + 64u * 512u);    // f
        const v16h b3 = *reinterpret_cast<const v16h*>(bptr + 96u * 512u);    // o
        bptr += (size_t)NT_ * 512;

        accI = __builtin_amdgcn_wmma_f32_16x16x32_f16(false, a.v, false, b0,
                                                      (short)0, accI, false, false);
        accG = __builtin_amdgcn_wmma_f32_16x16x32_f16(false, a.v, false, b1,
                                                      (short)0, accG, false, false);
        accF = __builtin_amdgcn_wmma_f32_16x16x32_f16(false, a.v, false, b2,
                                                      (short)0, accF, false, false);
        accO = __builtin_amdgcn_wmma_f32_16x16x32_f16(false, a.v, false, b3,
                                                      (short)0, accO, false, false);
    }

    // Fused LSTM cell update. C/D layout: lane column n = ng*16 + (lane&15);
    // VGPR v -> row m0 + v + (lane>=16 ? 8 : 0).
    const int n  = (ng << 4) + l16;
    const float bi = bias[n];
    const float bg = bias[H_ + n];
    const float bf = bias[2 * H_ + n];
    const float bo = bias[3 * H_ + n];

#pragma unroll
    for (int v = 0; v < 8; ++v) {
        const int m = m0 + v + (half ? 8 : 0);
        const float iv = sigmoidf_(accI[v] + bi);
        const float gv = tanhf(accG[v] + bg);
        const float fv = sigmoidf_(accF[v] + bf + 1.0f);
        const float ov = sigmoidf_(accO[v] + bo);
        const size_t cix = (size_t)m * H_ + n;
        const float cn = fv * c[cix] + iv * gv;
        const float hv = ov * tanhf(cn);
        c[cix] = cn;
        const _Float16 hh = (_Float16)hv;
        dA[(size_t)m * dA_stride + dA_off + n] = hh;
        if (dB)   dB[(size_t)m * dB_stride + dB_off + n] = hh;
        if (zdst) zdst[(size_t)m * H_ + n] = hv;
    }
}

// ---------------------------------------------------------------------------
// Head: m = z.Wm + bm (fed back into feat1 col 64), and per-AR-step outputs
//   mean = m * scale;  disp = softplus(z.Wd + bd) * sqrt(scale)
// One wave per batch row. Grid: 64 blocks x 256 threads.
// ---------------------------------------------------------------------------
__global__ __launch_bounds__(256) void head_kernel(
    const float* __restrict__ z,
    const float* __restrict__ Wm, const float* __restrict__ bm,
    const float* __restrict__ Wd, const float* __restrict__ bd,
    const float* __restrict__ scale,
    _Float16* __restrict__ feat1_next, float* __restrict__ out,
    int out_idx, int do_feed, int do_out)
{
    const int b    = blockIdx.x * (blockDim.x >> 5) + (threadIdx.x >> 5);
    const int lane = threadIdx.x & 31;
    float sm = 0.0f, sd = 0.0f;
    for (int k = lane; k < H_; k += 32) {
        const float zv = z[(size_t)b * H_ + k];
        sm += zv * Wm[k];
        sd += zv * Wd[k];
    }
#pragma unroll
    for (int off = 16; off; off >>= 1) {
        sm += __shfl_xor(sm, off, 32);
        sd += __shfl_xor(sd, off, 32);
    }
    if (lane == 0) {
        const float m = sm + bm[0];
        if (do_feed) feat1_next[(size_t)b * K1PAD + 64] = (_Float16)m;
        if (do_out) {
            const float s  = scale[b];
            const float dv = sd + bd[0];
            const float sp = (dv > 20.0f) ? dv : log1pf(__expf(dv));
            out[((size_t)b * (TP_ - TO_) + out_idx) * 2 + 0] = m * s;
            out[((size_t)b * (TP_ - TO_) + out_idx) * 2 + 1] = sp * sqrtf(s);
        }
    }
}

// ---------------------------------------------------------------------------
extern "C" void kernel_launch(void* const* d_in, const int* in_sizes, int n_in,
                              void* d_out, int out_size, void* d_ws, size_t ws_size,
                              hipStream_t stream)
{
    (void)in_sizes; (void)n_in; (void)out_size; (void)ws_size;
    const float* x  = (const float*)d_in[0];
    const float* y  = (const float*)d_in[1];
    const float* W1 = (const float*)d_in[2];
    const float* b1 = (const float*)d_in[3];
    const float* W2 = (const float*)d_in[4];
    const float* b2 = (const float*)d_in[5];
    const float* Wm = (const float*)d_in[6];
    const float* bm = (const float*)d_in[7];
    const float* Wd = (const float*)d_in[8];
    const float* bd = (const float*)d_in[9];
    float* out = (float*)d_out;
    char*  ws  = (char*)d_ws;

    // Workspace layout (all offsets 2 KB aligned)
    _Float16* W1h    = (_Float16*)(ws + 0);                   // 608*2048*2  = 2,490,368
    _Float16* W2h    = (_Float16*)(ws + 2490368);             // 1024*2048*2 = 4,194,304
    _Float16* feat1a = (_Float16*)(ws + 6684672);             // 512*608*2   =   622,592
    _Float16* feat1b = (_Float16*)(ws + 7307264);
    _Float16* feat2a = (_Float16*)(ws + 7929856);             // 512*1024*2  = 1,048,576
    _Float16* feat2b = (_Float16*)(ws + 8978432);
    float*    c1     = (float*)   (ws + 10027008);            // 512*512*4   = 1,048,576
    float*    c2     = (float*)   (ws + 11075584);
    float*    z      = (float*)   (ws + 12124160);
    float*    scale  = (float*)   (ws + 13172736);
    float*    invsc  = (float*)   (ws + 13174784);

    // Zero all recurrent state + feature buffers (feat1a .. c2, contiguous)
    hipMemsetAsync(ws + 6684672, 0, 12124160 - 6684672, stream);

    // One-time (per launch) weight conversion to WMMA-swizzled f16
    convert_weights_f16<<<KT1 * NT_, 32, 0, stream>>>(W1, F_ + 1 + H_, W1h);
    convert_weights_f16<<<KT2 * NT_, 32, 0, stream>>>(W2, 2 * H_, W2h);
    scale_kernel<<<B_ / 8, 256, 0, stream>>>(y, scale, invsc);

    for (int t = 0; t < TP_; ++t) {
        _Float16* f1r = (t & 1) ? feat1b : feat1a;   // read this step
        _Float16* f1w = (t & 1) ? feat1a : feat1b;   // h1 for next step
        _Float16* f2r = (t & 1) ? feat2b : feat2a;   // [h1_t | h2_{t-1}]
        _Float16* f2w = (t & 1) ? feat2a : feat2b;   // h2_t for next step

        prep_kernel<<<B_, F_, 0, stream>>>(x, y, invsc, f1r, t, t < TO_ ? 1 : 0);

        // Cell 1: K = 577 (pad 608). Writes h1 -> next feat1 (col 65+) and
        // this step's feat2 (col 0+).
        lstm_cell_wmma<<<128, 256, 0, stream>>>(
            f1r, K1PAD, KT1, W1h, b1, c1,
            f1w, K1PAD, F_ + 1,
            f2r, K2_, 0,
            nullptr);

        // Cell 2: K = 1024. Writes h2 -> next feat2 (col 512+) and fp32 z.
        lstm_cell_wmma<<<128, 256, 0, stream>>>(
            f2r, K2_, KT2, W2h, b2, c2,
            f2w, K2_, H_,
            nullptr, 0, 0,
            z);

        // Head: mean feedback for AR steps + outputs for decoded steps.
        if (t >= TO_ - 1) {
            head_kernel<<<B_ / 8, 256, 0, stream>>>(
                z, Wm, bm, Wd, bd, scale,
                f1w, out,
                t - TO_, (t < TP_ - 1) ? 1 : 0, (t >= TO_) ? 1 : 0);
        }
    }
}
